// SequentialPolicyNetwork_45878840656422
// MI455X (gfx1250) — matrix operand order repaired and checked by reading, compiled, not run on hardware
//
#include <hip/hip_runtime.h>
#include <math.h>

typedef __attribute__((ext_vector_type(16))) __bf16 v16bf;
typedef __attribute__((ext_vector_type(8)))  float  v8f;

union FragB { v16bf v; uint4 q[2]; };

#define HDIM 256
#define HPAD 264   // padded row length (bf16 elems) -> 528B row stride, conflict-free

__device__ __forceinline__ unsigned short f2bf(float f) {
    union { float f; unsigned u; } v; v.f = f;
    unsigned r = v.u + 0x7FFFu + ((v.u >> 16) & 1u);   // round-to-nearest-even
    return (unsigned short)(r >> 16);
}
__device__ __forceinline__ float bf2f(unsigned short h) {
    union { unsigned u; float f; } v; v.u = ((unsigned)h) << 16; return v.f;
}
__device__ __forceinline__ float sigm(float x) { return 1.f / (1.f + expf(-x)); }
__device__ __forceinline__ float gelu(float x) {
    return 0.5f * x * (1.f + erff(x * 0.70710678118654752f));
}

// ---------------- prep: encoder GEMM (state @ W_enc.T + b_enc) ----------------
__global__ void enc_gemm(const float* __restrict__ state, const float* __restrict__ Wenc,
                         const float* __restrict__ benc, float* __restrict__ emb) {
    __shared__ float srow[128];
    int b = blockIdx.x, t = threadIdx.x;
    if (t < 128) srow[t] = state[(size_t)b * 128 + t];
    __syncthreads();
    float s = benc[t];
    const float* wr = Wenc + (size_t)t * 128;
    for (int k = 0; k < 128; ++k) s += srow[k] * wr[k];
    emb[(size_t)b * 256 + t] = s;
}

// ---------------- prep: LayerNorm + GELU in place on emb ----------------
__global__ void enc_ln(float* __restrict__ emb, const float* __restrict__ g,
                       const float* __restrict__ be) {
    __shared__ float red[16];
    int b = blockIdx.x, t = threadIdx.x;
    float x = emb[(size_t)b * 256 + t];
    float s = x, sq = x * x;
    for (int m = 16; m >= 1; m >>= 1) { s += __shfl_xor(s, m, 32); sq += __shfl_xor(sq, m, 32); }
    int wv = t >> 5, ln = t & 31;
    if (ln == 0) { red[wv] = s; red[8 + wv] = sq; }
    __syncthreads();
    float ts = 0.f, tq = 0.f;
    for (int i = 0; i < 8; ++i) { ts += red[i]; tq += red[8 + i]; }
    float mean = ts * (1.f / 256.f);
    float var  = tq * (1.f / 256.f) - mean * mean;
    float inv  = rsqrtf(var + 1e-5f);
    float xn = (x - mean) * inv * g[t] + be[t];
    emb[(size_t)b * 256 + t] = gelu(xn);
}

// ---- prep: z0_static = emb @ Wih0[:, :256].T + bih0 + bhh0, in C-frag layout ----
__global__ void zstatic_k(const float* __restrict__ emb, const float* __restrict__ Wih0,
                          const float* __restrict__ bih0, const float* __restrict__ bhh0,
                          float* __restrict__ zs, int total) {
    int t = blockIdx.x * 256 + threadIdx.x;
    if (t >= total) return;
    int b = t >> 10, n = t & 1023;
    const float* e = emb + (size_t)b * 256;
    const float* w = Wih0 + (size_t)n * 260;
    float s = bih0[n] + bhh0[n];
    for (int k = 0; k < 256; ++k) s += e[k] * w[k];
    int mt = b >> 4, mr = b & 15;
    int nt = n >> 4, nl = n & 15;
    int hi = mr >> 3, j = mr & 7;
    int lane = nl + 16 * hi;
    zs[(((size_t)mt * 64 + nt) * 32 + lane) * 8 + j] = s;
}

// ---- prep: repack f32 weight [N,ldw] rows -> bf16 WMMA B-fragments ----
// B-frag (32x16 bf16): lane l holds column N = nt*16 + (l&15), K = kt*32 + (l>>4)*8 + (0..7), then +16 + (0..7)
__global__ void repack_b(const float* __restrict__ W, int ldw, int col0, int kvalid,
                         int KT, unsigned short* __restrict__ dst) {
    int fi = blockIdx.x;               // nt*KT + kt
    int nt = fi / KT, kt = fi % KT;
    int l = threadIdx.x;               // 0..31
    int r = nt * 16 + (l & 15);
    int k0 = kt * 32 + (l >> 4) * 8;
    __attribute__((aligned(16))) unsigned short o[16];
    for (int e = 0; e < 16; ++e) {
        int k = k0 + (e & 7) + (e >> 3) * 16;
        float v = (k < kvalid) ? W[(size_t)r * ldw + col0 + k] : 0.f;
        o[e] = f2bf(v);
    }
    unsigned short* p = dst + (((size_t)fi * 32 + l) * 16);
    *(uint4*)p = *(const uint4*)o;
    *(uint4*)(p + 8) = *(const uint4*)(o + 8);
}

__global__ void bias_sum(const float* __restrict__ a, const float* __restrict__ b,
                         float* __restrict__ o, int n) {
    int i = blockIdx.x * 256 + threadIdx.x;
    if (i < n) o[i] = a[i] + b[i];
}

// ---------------- main persistent recurrent kernel ----------------
// 512 threads = 16 wave32s. Wave w owns gate-column slice [w*16, (w+1)*16):
// one 16x16 WMMA tile per gate (4 acc tiles), c-state one v8f per layer.
// `opq` is 0 at runtime; `zo = step & opq` defeats LICM of the weight streams
// (round-2 showed the compiler hoisting loop-invariant weight loads and
// spilling them to per-lane scratch -> private L2-thrashing copies).
__global__ __launch_bounds__(512) void policy_main(
    const int* __restrict__ seqp, int B, int opq,
    const float* __restrict__ zs,
    const unsigned short* __restrict__ whh0f, const unsigned short* __restrict__ waf,
    const unsigned short* __restrict__ wih1f, const unsigned short* __restrict__ whh1f,
    const unsigned short* __restrict__ wd1f,  const float* __restrict__ bias1,
    const float* __restrict__ bd1, const float* __restrict__ g_dec,
    const float* __restrict__ beta_dec, const float* __restrict__ Wd2,
    const float* __restrict__ bd2, float* __restrict__ out)
{
    __shared__ unsigned short h0b[16 * HPAD];
    __shared__ unsigned short h1b[16 * HPAD];
    __shared__ unsigned short actb[16 * 32];
    __shared__ float          dbuf[16 * HPAD];
    __shared__ unsigned short decb[16 * HPAD];

    const int tid = threadIdx.x;
    const int w   = tid >> 5;        // wave id 0..15 (wave32)
    const int l   = tid & 31;
    const int hi  = l >> 4;
    const int nl  = l & 15;
    const int mt  = blockIdx.x;
    const int m0  = mt * 16;

    const int seq = seqp[0];
    const size_t BH   = (size_t)B * HDIM;
    const size_t OFFH = (size_t)B * seq * 4;
    const size_t OFFC = OFFH + 2 * BH;

    for (int i = tid; i < 16 * HPAD; i += 512) { h0b[i] = 0; h1b[i] = 0; }
    for (int i = tid; i < 16 * 32; i += 512) actb[i] = 0;
    __syncthreads();

    v8f zero8; for (int e = 0; e < 8; ++e) zero8[e] = 0.f;
    v8f c0 = zero8, c1 = zero8;
    v8f acc[4];
    const int col = w * 16 + nl;      // this wave's hidden-state column

    for (int step = 0; step < seq; ++step) {
        const bool last = (step == seq - 1);
        const int  zo   = step & opq;   // == 0, but not provably so: blocks LICM

        // ================= LSTM layer 0 =================
        // init acc tiles from precomputed z0_static fragments
        #pragma unroll
        for (int g = 0; g < 4; ++g) {
            int nt = g * 16 + w;
            const float4* p = (const float4*)(zs + (((size_t)mt * 64 + nt) * 32 + l) * 8 + zo);
            float4 q0 = p[0], q1 = p[1];
            v8f a; a[0]=q0.x; a[1]=q0.y; a[2]=q0.z; a[3]=q0.w;
                   a[4]=q1.x; a[5]=q1.y; a[6]=q1.z; a[7]=q1.w;
            acc[g] = a;
        }
        // h0 @ Whh0.T : batch the 4 B-fragment loads, then 4 WMMAs
        for (int kt = 0; kt < 8; ++kt) {
            FragB a;
            const unsigned short* pa = &h0b[nl * HPAD + kt * 32 + hi * 8];
            a.q[0] = *(const uint4*)pa; a.q[1] = *(const uint4*)(pa + 16);
            FragB b[4];
            #pragma unroll
            for (int g = 0; g < 4; ++g) {
                const unsigned short* pb =
                    whh0f + (((size_t)((g * 16 + w) * 8 + kt)) * 32 + l) * 16 + zo;
                b[g].q[0] = *(const uint4*)pb; b[g].q[1] = *(const uint4*)(pb + 8);
            }
            #pragma unroll
            for (int g = 0; g < 4; ++g)
                acc[g] = __builtin_amdgcn_wmma_f32_16x16x32_bf16(
                    false, a.v, false, b[g].v, (short)0, acc[g], false, false);
        }
        // action contribution (K padded to 32)
        {
            FragB a;
            const unsigned short* pa = &actb[nl * 32 + hi * 8];
            a.q[0] = *(const uint4*)pa; a.q[1] = *(const uint4*)(pa + 16);
            FragB b[4];
            #pragma unroll
            for (int g = 0; g < 4; ++g) {
                const unsigned short* pb = waf + (((size_t)(g * 16 + w)) * 32 + l) * 16 + zo;
                b[g].q[0] = *(const uint4*)pb; b[g].q[1] = *(const uint4*)(pb + 8);
            }
            #pragma unroll
            for (int g = 0; g < 4; ++g)
                acc[g] = __builtin_amdgcn_wmma_f32_16x16x32_bf16(
                    false, a.v, false, b[g].v, (short)0, acc[g], false, false);
        }
        __syncthreads();   // all reads of h0b/actb done

        // gates (i,f,g,o wave-local)
        #pragma unroll
        for (int e = 0; e < 8; ++e) {
            float iv = sigm(acc[0][e]);
            float fv = sigm(acc[1][e]);
            float gv = tanhf(acc[2][e]);
            float ov = sigm(acc[3][e]);
            float cn = fv * c0[e] + iv * gv;
            float hn = ov * tanhf(cn);
            c0[e] = cn;
            int m = e + 8 * hi;
            h0b[m * HPAD + col] = f2bf(hn);
            if (last) {
                out[OFFH + (size_t)(m0 + m) * HDIM + col] = hn;
                out[OFFC + (size_t)(m0 + m) * HDIM + col] = cn;
            }
        }
        __syncthreads();

        // ================= LSTM layer 1 =================
        #pragma unroll
        for (int g = 0; g < 4; ++g) {
            float bv = bias1[(g * 16 + w) * 16 + nl];
            v8f a; for (int e = 0; e < 8; ++e) a[e] = bv;
            acc[g] = a;
        }
        // pass 1: h0 @ Wih1.T
        for (int kt = 0; kt < 8; ++kt) {
            FragB a;
            const unsigned short* pa = &h0b[nl * HPAD + kt * 32 + hi * 8];
            a.q[0] = *(const uint4*)pa; a.q[1] = *(const uint4*)(pa + 16);
            FragB b[4];
            #pragma unroll
            for (int g = 0; g < 4; ++g) {
                const unsigned short* pb =
                    wih1f + (((size_t)((g * 16 + w) * 8 + kt)) * 32 + l) * 16 + zo;
                b[g].q[0] = *(const uint4*)pb; b[g].q[1] = *(const uint4*)(pb + 8);
            }
            #pragma unroll
            for (int g = 0; g < 4; ++g)
                acc[g] = __builtin_amdgcn_wmma_f32_16x16x32_bf16(
                    false, a.v, false, b[g].v, (short)0, acc[g], false, false);
        }
        // pass 2: h1 @ Whh1.T
        for (int kt = 0; kt < 8; ++kt) {
            FragB a;
            const unsigned short* pa = &h1b[nl * HPAD + kt * 32 + hi * 8];
            a.q[0] = *(const uint4*)pa; a.q[1] = *(const uint4*)(pa + 16);
            FragB b[4];
            #pragma unroll
            for (int g = 0; g < 4; ++g) {
                const unsigned short* pb =
                    whh1f + (((size_t)((g * 16 + w) * 8 + kt)) * 32 + l) * 16 + zo;
                b[g].q[0] = *(const uint4*)pb; b[g].q[1] = *(const uint4*)(pb + 8);
            }
            #pragma unroll
            for (int g = 0; g < 4; ++g)
                acc[g] = __builtin_amdgcn_wmma_f32_16x16x32_bf16(
                    false, a.v, false, b[g].v, (short)0, acc[g], false, false);
        }
        __syncthreads();
        #pragma unroll
        for (int e = 0; e < 8; ++e) {
            float iv = sigm(acc[0][e]);
            float fv = sigm(acc[1][e]);
            float gv = tanhf(acc[2][e]);
            float ov = sigm(acc[3][e]);
            float cn = fv * c1[e] + iv * gv;
            float hn = ov * tanhf(cn);
            c1[e] = cn;
            int m = e + 8 * hi;
            h1b[m * HPAD + col] = f2bf(hn);
            if (last) {
                out[OFFH + BH + (size_t)(m0 + m) * HDIM + col] = hn;
                out[OFFC + BH + (size_t)(m0 + m) * HDIM + col] = cn;
            }
        }
        __syncthreads();

        // ================= decoder: d = h1 @ Wd1.T + bd1 (one tile per wave) =====
        {
            float bv = bd1[col];
            v8f dacc; for (int e = 0; e < 8; ++e) dacc[e] = bv;
            for (int kt = 0; kt < 8; ++kt) {
                FragB a;
                const unsigned short* pa = &h1b[nl * HPAD + kt * 32 + hi * 8];
                a.q[0] = *(const uint4*)pa; a.q[1] = *(const uint4*)(pa + 16);
                FragB b;
                const unsigned short* pb = wd1f + (((size_t)(w * 8 + kt)) * 32 + l) * 16 + zo;
                b.q[0] = *(const uint4*)pb; b.q[1] = *(const uint4*)(pb + 8);
                dacc = __builtin_amdgcn_wmma_f32_16x16x32_bf16(
                    false, a.v, false, b.v, (short)0, dacc, false, false);
            }
            #pragma unroll
            for (int e = 0; e < 8; ++e)
                dbuf[(e + 8 * hi) * HPAD + col] = dacc[e];
        }
        __syncthreads();

        // LayerNorm + GELU over rows of d; wave w handles row w
        {
            float s = 0.f, sq = 0.f;
            float xv[8];
            #pragma unroll
            for (int t = 0; t < 8; ++t) {
                float x = dbuf[w * HPAD + l + t * 32];
                xv[t] = x; s += x; sq += x * x;
            }
            for (int m = 16; m >= 1; m >>= 1) { s += __shfl_xor(s, m, 32); sq += __shfl_xor(sq, m, 32); }
            float mean = s * (1.f / 256.f);
            float var  = sq * (1.f / 256.f) - mean * mean;
            float inv  = rsqrtf(var + 1e-5f);
            #pragma unroll
            for (int t = 0; t < 8; ++t) {
                int c = l + t * 32;
                float x = (xv[t] - mean) * inv * g_dec[c] + beta_dec[c];
                decb[w * HPAD + c] = f2bf(gelu(x));
            }
        }
        __syncthreads();

        // final: a = dec @ Wd2.T + bd2  (tiny: waves 0,1 only)
        if (w < 2) {
            int m = nl;
            int aidx = hi + 2 * w;
            float s2 = bd2[aidx];
            const float* wr = Wd2 + (size_t)aidx * 256 + zo;
            for (int k = 0; k < 256; ++k) s2 += bf2f(decb[m * HPAD + k]) * wr[k];
            out[((size_t)(m0 + m) * seq + step) * 4 + aidx] = s2;
            actb[m * 32 + aidx] = f2bf(s2);
        }
        __syncthreads();
    }
}

extern "C" void kernel_launch(void* const* d_in, const int* in_sizes, int n_in,
                              void* d_out, int out_size, void* d_ws, size_t ws_size,
                              hipStream_t stream) {
    const float* state  = (const float*)d_in[0];
    const int*   seqp   = (const int*)  d_in[1];
    const float* Wenc   = (const float*)d_in[2];
    const float* benc   = (const float*)d_in[3];
    const float* g_enc  = (const float*)d_in[4];
    const float* b_encln= (const float*)d_in[5];
    const float* Wih0   = (const float*)d_in[6];
    const float* Whh0   = (const float*)d_in[7];
    const float* bih0   = (const float*)d_in[8];
    const float* bhh0   = (const float*)d_in[9];
    const float* Wih1   = (const float*)d_in[10];
    const float* Whh1   = (const float*)d_in[11];
    const float* bih1   = (const float*)d_in[12];
    const float* bhh1   = (const float*)d_in[13];
    const float* Wd1    = (const float*)d_in[14];
    const float* bd1    = (const float*)d_in[15];
    const float* g_dec  = (const float*)d_in[16];
    const float* beta_dec=(const float*)d_in[17];
    const float* Wd2    = (const float*)d_in[18];
    const float* bd2    = (const float*)d_in[19];
    float* out = (float*)d_out;

    const int B = in_sizes[0] / 128;

    // ---- carve workspace ----
    char* p = (char*)d_ws;
    float* emb = (float*)p;                 p += (size_t)B * 256 * 4;
    float* zs  = (float*)p;                 p += (size_t)B * 1024 * 4;
    unsigned short* whh0f = (unsigned short*)p; p += (size_t)64 * 8 * 32 * 16 * 2;
    unsigned short* wih1f = (unsigned short*)p; p += (size_t)64 * 8 * 32 * 16 * 2;
    unsigned short* whh1f = (unsigned short*)p; p += (size_t)64 * 8 * 32 * 16 * 2;
    unsigned short* wd1f  = (unsigned short*)p; p += (size_t)16 * 8 * 32 * 16 * 2;
    unsigned short* waf   = (unsigned short*)p; p += (size_t)64 * 1 * 32 * 16 * 2;
    float* bias1v = (float*)p;              p += 1024 * 4;

    // ---- prep (run once per call; deterministic) ----
    enc_gemm<<<B, 256, 0, stream>>>(state, Wenc, benc, emb);
    enc_ln<<<B, 256, 0, stream>>>(emb, g_enc, b_encln);
    zstatic_k<<<B * 4, 256, 0, stream>>>(emb, Wih0, bih0, bhh0, zs, B * 1024);
    repack_b<<<64 * 8, 32, 0, stream>>>(Whh0, 256, 0, 256, 8, whh0f);
    repack_b<<<64 * 8, 32, 0, stream>>>(Wih1, 256, 0, 256, 8, wih1f);
    repack_b<<<64 * 8, 32, 0, stream>>>(Whh1, 256, 0, 256, 8, whh1f);
    repack_b<<<16 * 8, 32, 0, stream>>>(Wd1, 256, 0, 256, 8, wd1f);
    repack_b<<<64 * 1, 32, 0, stream>>>(Wih0, 260, 256, 4, 1, waf);
    bias_sum<<<4, 256, 0, stream>>>(bih1, bhh1, bias1v, 1024);

    // ---- main recurrent kernel: one 16-row batch tile per workgroup, 16 waves ----
    policy_main<<<B / 16, 512, 0, stream>>>(seqp, B, /*opq=*/0, zs, whh0f, waf,
                                            wih1f, whh1f, wd1f, bias1v, bd1,
                                            g_dec, beta_dec, Wd2, bd2, out);
}
